// OptimalTransportLoss_76510547411095
// MI455X (gfx1250) — matrix operand order repaired and checked by reading, not compiled, not run on hardware
//
#include <hip/hip_runtime.h>
#include <stdint.h>

// ---------------- problem constants (reference: B=4096, C=1000, reg=0.1, 100 iters)
#define BATCH     4096
#define NCLS      1000
#define REG_INV   10.0f          // 1/reg
#define REG_F     0.1f
#define N_ITERS   100
#define JS        8              // j-split for GEMV partials
#define ITILES    256            // 4096/16  (M tiles)
#define JTILES    128            // 4096/32  (K tiles)
#define NTILES    (ITILES*JTILES)       // 32768 tiles per matrix copy
#define BUILD_THREADS (NTILES*64)       // each thread emits one uint4 (8 bf16)

typedef __attribute__((ext_vector_type(16))) __bf16 v16bf;
typedef __attribute__((ext_vector_type(8)))  float  v8f;

union BfPack { uint32_t u[8]; uint4 q[2]; v16bf v; };

__device__ __forceinline__ uint32_t f2bf_rne(float x) {
  union { float f; uint32_t u; } c; c.f = x;
  return (c.u + 0x7FFFu + ((c.u >> 16) & 1u)) >> 16;   // round-to-nearest-even bf16
}
__device__ __forceinline__ float bf2f(uint32_t hbits) {
  union { uint32_t u; float f; } c; c.u = hbits << 16;
  return c.f;
}

// ---------------- init: v = 1 (f32 and pre-packed bf16 image)
__global__ void ot_init_v(float* __restrict__ v, uint16_t* __restrict__ vbf) {
  int i = blockIdx.x * blockDim.x + threadIdx.x;
  if (i < BATCH) { v[i] = 1.0f; vbf[i] = 0x3F80u; }
}

// ---------------- per-row softmax stats (max, 1/Z, sum p^2)
__global__ void ot_rowstats(const float* __restrict__ preds, float4* __restrict__ rs) {
  __shared__ float r0[256];
  __shared__ float r1[256];
  const int row = blockIdx.x;
  const int tid = threadIdx.x;
  const float* x = preds + row * NCLS;

  float m = -3.4e38f;
  for (int c = tid; c < NCLS; c += 256) m = fmaxf(m, x[c]);
  r0[tid] = m; __syncthreads();
  for (int s = 128; s > 0; s >>= 1) {
    if (tid < s) r0[tid] = fmaxf(r0[tid], r0[tid + s]);
    __syncthreads();
  }
  m = r0[0]; __syncthreads();

  float s1 = 0.0f, s2 = 0.0f;
  for (int c = tid; c < NCLS; c += 256) {
    float e = __expf(x[c] - m);
    s1 += e; s2 += e * e;
  }
  r0[tid] = s1; r1[tid] = s2; __syncthreads();
  for (int s = 128; s > 0; s >>= 1) {
    if (tid < s) { r0[tid] += r0[tid + s]; r1[tid] += r1[tid + s]; }
    __syncthreads();
  }
  if (tid == 0) {
    float Z = r0[0];
    float4 o; o.x = m; o.y = 1.0f / Z; o.z = r1[0] / (Z * Z); o.w = 0.0f;
    rs[row] = o;
  }
}

// K_ij = exp(-C_ij/reg),  C_ij = sqrt(max(sq_i + 1 - 2*p[i,targets[j]], 0))
__device__ __forceinline__ float kelem(const float* __restrict__ preds,
                                       const int*   __restrict__ targets,
                                       int i, int j, float4 st) {
  int t = targets[j];
  float pv = __expf(preds[i * NCLS + t] - st.x) * st.y;
  float d  = st.z + 1.0f - 2.0f * pv;
  float c  = sqrtf(fmaxf(d, 0.0f));
  return __expf(-REG_INV * c);
}

// -------- build K (and K^T) in bf16, pre-swizzled into the WMMA 16x32 A-layout.
// Tile = it*JTILES + jt (1KB each). The uint4 at sub = h*32 + lane is what lane
// `lane` loads into A VGPRs [h*4 .. h*4+3]: M = lane%16, K = h*16+(lane/16)*8+e.
__global__ void ot_buildK(const float* __restrict__ preds,
                          const int*   __restrict__ targets,
                          const float4* __restrict__ rs,
                          uint4* __restrict__ Kt, int transposed) {
  int t = blockIdx.x * blockDim.x + threadIdx.x;       // 0 .. BUILD_THREADS-1
  if (t >= BUILD_THREADS) return;
  int tile = t >> 6, sub = t & 63;
  int l = sub & 31, h = sub >> 5;
  int m  = l & 15;
  int kb = h * 16 + ((l >> 4) << 3);

  uint32_t out[4];
  if (!transposed) {                 // rows = i (M), reduce-dim = j (K)
    int it = tile >> 7, jt = tile & 127;
    int i  = it * 16 + m;
    int jb = jt * 32 + kb;
    float4 st = rs[i];
#pragma unroll
    for (int p = 0; p < 4; ++p) {
      float k0 = kelem(preds, targets, i, jb + 2 * p,     st);
      float k1 = kelem(preds, targets, i, jb + 2 * p + 1, st);
      out[p] = (f2bf_rne(k1) << 16) | f2bf_rne(k0);
    }
  } else {                           // rows = j (M), reduce-dim = i (K)
    int jt2 = tile >> 7, it2 = tile & 127;
    int j  = jt2 * 16 + m;
    int ib = it2 * 32 + kb;
#pragma unroll
    for (int p = 0; p < 4; ++p) {
      int i0 = ib + 2 * p, i1 = ib + 2 * p + 1;
      float k0 = kelem(preds, targets, i0, j, rs[i0]);
      float k1 = kelem(preds, targets, i1, j, rs[i1]);
      out[p] = (f2bf_rne(k1) << 16) | f2bf_rne(k0);
    }
  }
  uint4 q; q.x = out[0]; q.y = out[1]; q.z = out[2]; q.w = out[3];
  Kt[t] = q;
}

// -------- WMMA GEMV:  part[js][row] = sum over this wave's j-slice of K[row,j]*vec[j]
// One wave per (row-tile, j-slice). A = pre-swizzled 16x32 bf16 tile (2x B128),
// B = pre-packed bf16 vector slice (2x B128, wave-broadcast), D accumulates.
// Inner loop = 4 loads + 1 WMMA; two accumulators break the SRC2 chain.
__global__ void ot_gemv(const uint4* __restrict__ Kt,
                        const uint4* __restrict__ vecbf,   // BATCH bf16, packed
                        float* __restrict__ part) {
  const int lane = threadIdx.x & 31;
  const int w    = blockIdx.x * (blockDim.x >> 5) + (threadIdx.x >> 5);
  const int it   = w >> 3;          // 0..255
  const int js   = w & (JS - 1);    // 0..7

  v8f acc0 = {};
  v8f acc1 = {};
  const int jt0   = js * (JTILES / JS);
  const int bhalf = lane >> 4;          // +0 or +1 uint4 within the 32-elem slice

#pragma unroll 8
  for (int jj = 0; jj < JTILES / JS; jj += 2) {
    int jt   = jt0 + jj;
    int tb   = (it * JTILES + jt) * 64;
    // ---- tile jt (even) -> acc0
    BfPack A0, B0;
    A0.q[0] = Kt[tb + lane];
    A0.q[1] = Kt[tb + 32 + lane];
    B0.q[0] = vecbf[jt * 4 + bhalf];
    B0.q[1] = vecbf[jt * 4 + bhalf + 2];
    acc0 = __builtin_amdgcn_wmma_f32_16x16x32_bf16(
        false, A0.v, false, B0.v, (short)0, acc0, false, false);
    // ---- tile jt+1 (odd) -> acc1
    BfPack A1, B1;
    A1.q[0] = Kt[tb + 64 + lane];
    A1.q[1] = Kt[tb + 96 + lane];
    B1.q[0] = vecbf[jt * 4 + 4 + bhalf];
    B1.q[1] = vecbf[jt * 4 + 4 + bhalf + 2];
    acc1 = __builtin_amdgcn_wmma_f32_16x16x32_bf16(
        false, A1.v, false, B1.v, (short)0, acc1, false, false);
  }
  v8f acc = acc0 + acc1;

  // D layout: VGPR r -> (M=r, N=lane) for lanes 0-15, (M=8+r, N=lane-16) for 16-31
  float* dst = part + js * BATCH + it * 16;
  if (lane == 0) {
#pragma unroll
    for (int r = 0; r < 8; ++r) dst[r] = acc[r];
  } else if (lane == 16) {
#pragma unroll
    for (int r = 0; r < 8; ++r) dst[8 + r] = acc[r];
  }
}

// -------- finalize: out[i] = (1/B) / sum_js part[js][i]; also emit bf16 image
__global__ void ot_fin(const float* __restrict__ part,
                       float* __restrict__ out, uint16_t* __restrict__ outbf) {
  int i = blockIdx.x * blockDim.x + threadIdx.x;
  if (i >= BATCH) return;
  float s = 0.0f;
#pragma unroll
  for (int js = 0; js < JS; ++js) s += part[js * BATCH + i];
  float val = (1.0f / (float)BATCH) / s;
  out[i]   = val;
  outbf[i] = (uint16_t)f2bf_rne(val);
}

// -------- loss stage 1: sum_ij u_i * K_ij * v_j * (-reg * ln K_ij), block partials
__global__ void ot_loss1(const uint4* __restrict__ Kt1,
                         const float* __restrict__ u,
                         const float* __restrict__ v,
                         float* __restrict__ bpart) {
  __shared__ float red[256];
  int t = blockIdx.x * blockDim.x + threadIdx.x;     // one uint4 (8 elems) / thread
  float s = 0.0f;
  if (t < BUILD_THREADS) {
    uint4 q = Kt1[t];
    int tile = t >> 6, sub = t & 63;
    int l = sub & 31, h = sub >> 5;
    int it = tile >> 7, jt = tile & 127;
    int i  = it * 16 + (l & 15);
    int jb = jt * 32 + h * 16 + ((l >> 4) << 3);
    float ui = u[i];
    uint32_t uw[4] = { q.x, q.y, q.z, q.w };
#pragma unroll
    for (int e = 0; e < 8; ++e) {
      float k = bf2f((uw[e >> 1] >> ((e & 1) * 16)) & 0xFFFFu);
      float c = -REG_F * __logf(k);
      s += ui * v[jb + e] * k * c;
    }
  }
  red[threadIdx.x] = s; __syncthreads();
  for (int sft = 128; sft > 0; sft >>= 1) {
    if (threadIdx.x < sft) red[threadIdx.x] += red[threadIdx.x + sft];
    __syncthreads();
  }
  if (threadIdx.x == 0) bpart[blockIdx.x] = red[0];
}

// -------- loss stage 2: deterministic final reduce -> d_out[0]
__global__ void ot_loss2(const float* __restrict__ bpart, int n, float* __restrict__ out) {
  __shared__ float red[256];
  float s = 0.0f;
  for (int i = threadIdx.x; i < n; i += 256) s += bpart[i];
  red[threadIdx.x] = s; __syncthreads();
  for (int sft = 128; sft > 0; sft >>= 1) {
    if (threadIdx.x < sft) red[threadIdx.x] += red[threadIdx.x + sft];
    __syncthreads();
  }
  if (threadIdx.x == 0) out[0] = red[0];
}

extern "C" void kernel_launch(void* const* d_in, const int* in_sizes, int n_in,
                              void* d_out, int out_size, void* d_ws, size_t ws_size,
                              hipStream_t stream) {
  const float* preds   = (const float*)d_in[0];
  const int*   targets = (const int*)d_in[1];

  char* ws = (char*)d_ws;
  size_t off = 0;
  uint4*    Kt1 = (uint4*)(ws + off);    off += (size_t)BATCH * BATCH * 2;  // 32 MB
  uint4*    Kt2 = (uint4*)(ws + off);    off += (size_t)BATCH * BATCH * 2;  // 32 MB
  float4*   rs  = (float4*)(ws + off);   off += (size_t)BATCH * 16;
  float*    u   = (float*)(ws + off);    off += (size_t)BATCH * 4;
  float*    v   = (float*)(ws + off);    off += (size_t)BATCH * 4;
  uint16_t* ubf = (uint16_t*)(ws + off); off += (size_t)BATCH * 2;
  uint16_t* vbf = (uint16_t*)(ws + off); off += (size_t)BATCH * 2;
  float*    part= (float*)(ws + off);    off += (size_t)JS * BATCH * 4;
  float*    lp  = (float*)(ws + off);    off += (size_t)8192 * 4;

  // init v = 1 (u computed first); no other state carried across calls
  ot_init_v<<<(BATCH + 255) / 256, 256, 0, stream>>>(v, vbf);

  // softmax row stats
  ot_rowstats<<<BATCH, 256, 0, stream>>>(preds, rs);

  // build K and K^T, bf16, pre-swizzled to WMMA A-layout
  ot_buildK<<<BUILD_THREADS / 256, 256, 0, stream>>>(preds, targets, rs, Kt1, 0);
  ot_buildK<<<BUILD_THREADS / 256, 256, 0, stream>>>(preds, targets, rs, Kt2, 1);

  // 100 Sinkhorn scaling iterations: u = a/(K v); v = a/(K^T u)
  const int gemvBlocks = (ITILES * JS) / 4;   // 4 waves per 128-thread block
  for (int it = 0; it < N_ITERS; ++it) {
    ot_gemv<<<gemvBlocks, 128, 0, stream>>>(Kt1, (const uint4*)vbf, part);
    ot_fin <<<(BATCH + 255) / 256, 256, 0, stream>>>(part, u, ubf);
    ot_gemv<<<gemvBlocks, 128, 0, stream>>>(Kt2, (const uint4*)ubf, part);
    ot_fin <<<(BATCH + 255) / 256, 256, 0, stream>>>(part, v, vbf);
  }

  // loss = sum u_i K_ij v_j * C_ij with C = -reg*ln(K)
  int nblk = BUILD_THREADS / 256;             // 8192
  ot_loss1<<<nblk, 256, 0, stream>>>(Kt1, u, v, lp);
  ot_loss2<<<1, 256, 0, stream>>>(lp, nblk, (float*)d_out);
}
